// VisionMambaPredictor_30975304138902
// MI455X (gfx1250) — matrix operand order repaired and checked by reading, not compiled, not run on hardware
//
#include <hip/hip_runtime.h>
#include <stddef.h>

// ---------------------------------------------------------------------------
// Types for CDNA5 WMMA (wave32)
// ---------------------------------------------------------------------------
typedef __bf16 bf16_t;
typedef __attribute__((ext_vector_type(16))) __bf16 v16bf;
typedef __attribute__((ext_vector_type(8)))  __bf16 v8bf;
typedef __attribute__((ext_vector_type(8)))  float  v8f;

#define DEPTH_   6
#define DM_      768      // d_model
#define DI_      1536     // d_inner
#define DS_      16       // d_state
#define RK_      48       // dt_rank
#define RKP_     64       // dt_rank padded to mult of 32
#define XP_      80       // dt_rank + 2*d_state
#define XPP_     96       // padded to mult of 32
#define BATCH_   2
#define L_       784      // sequence length (N_CTXT + N_TGT)
#define NC_      392
#define T_       (BATCH_*L_)   // 1568 total token rows

__device__ __forceinline__ v16bf cat8(v8bf lo, v8bf hi) {
  return __builtin_shufflevector(lo, hi, 0,1,2,3,4,5,6,7,8,9,10,11,12,13,14,15);
}
__device__ __forceinline__ bf16_t f2bf(float f) { return (bf16_t)f; }

// ---------------------------------------------------------------------------
// Generic bf16 GEMM:  C(f32, MxN) = A(bf16, MxK, row major) @ W(bf16, NxK)^T
// One wave -> 32x32 output tile via 4x v_wmma_f32_16x16x32_bf16 per K-step.
// Requires K % 32 == 0 (enforced by padding on the host side).
// ---------------------------------------------------------------------------
__global__ void gemm_bf16_kernel(const bf16_t* __restrict__ A,
                                 const bf16_t* __restrict__ W,
                                 float* __restrict__ C,
                                 int M, int N, int K) {
  const int wave = (blockIdx.x * blockDim.x + threadIdx.x) >> 5;
  const int lane = threadIdx.x & 31;
  const int mt = (M + 31) >> 5;
  const int nt = (N + 31) >> 5;
  if (wave >= mt * nt) return;           // wave-uniform exit, EXEC stays full
  const int m0 = (wave / nt) * 32;
  const int n0 = (wave % nt) * 32;
  const int hl = lane >> 4;              // half-wave select
  const int ll = lane & 15;

  int rA0 = m0 + ll;        if (rA0 > M - 1) rA0 = M - 1;
  int rA1 = m0 + 16 + ll;   if (rA1 > M - 1) rA1 = M - 1;
  int cB0 = n0 + ll;        if (cB0 > N - 1) cB0 = N - 1;
  int cB1 = n0 + 16 + ll;   if (cB1 > N - 1) cB1 = N - 1;

  // A frag (16x32 bf16): lane ll, half hl -> K chunks {hl*8..+7} and {16+hl*8..+7}
  const bf16_t* pa0 = A + (size_t)rA0 * K + hl * 8;
  const bf16_t* pa1 = A + (size_t)rA1 * K + hl * 8;
  // B frag (32x16 bf16) = W^T: lane ll holds column N, K chunks {hl*8..+7} and {16+hl*8..+7}
  const bf16_t* pb0 = W + (size_t)cB0 * K + hl * 8;
  const bf16_t* pb1 = W + (size_t)cB1 * K + hl * 8;

  v8f acc00 = {}, acc01 = {}, acc10 = {}, acc11 = {};
  for (int kk = 0; kk < K; kk += 32) {
    __builtin_prefetch(pa0 + kk + 128, 0, 3);   // global_prefetch_b8
    __builtin_prefetch(pb0 + kk + 128, 0, 3);
    v16bf a0 = cat8(*(const v8bf*)(pa0 + kk), *(const v8bf*)(pa0 + kk + 16));
    v16bf a1 = cat8(*(const v8bf*)(pa1 + kk), *(const v8bf*)(pa1 + kk + 16));
    v16bf b0 = cat8(*(const v8bf*)(pb0 + kk), *(const v8bf*)(pb0 + kk + 16));
    v16bf b1 = cat8(*(const v8bf*)(pb1 + kk), *(const v8bf*)(pb1 + kk + 16));
    acc00 = __builtin_amdgcn_wmma_f32_16x16x32_bf16(false, a0, false, b0, (short)0, acc00, false, false);
    acc01 = __builtin_amdgcn_wmma_f32_16x16x32_bf16(false, a0, false, b1, (short)0, acc01, false, false);
    acc10 = __builtin_amdgcn_wmma_f32_16x16x32_bf16(false, a1, false, b0, (short)0, acc10, false, false);
    acc11 = __builtin_amdgcn_wmma_f32_16x16x32_bf16(false, a1, false, b1, (short)0, acc11, false, false);
  }

  // C/D layout: lanes 0-15: N=ll, VGPR v -> M=v ; lanes 16-31: N=ll, VGPR v -> M=8+v
  const int col0 = n0 + ll, col1 = n0 + 16 + ll;
  const int rb0 = m0 + hl * 8, rb1 = m0 + 16 + hl * 8;
#pragma unroll
  for (int v = 0; v < 8; ++v) {
    int r0 = rb0 + v, r1 = rb1 + v;
    if (r0 < M) {
      if (col0 < N) C[(size_t)r0 * N + col0] = acc00[v];
      if (col1 < N) C[(size_t)r0 * N + col1] = acc01[v];
    }
    if (r1 < M) {
      if (col0 < N) C[(size_t)r1 * N + col0] = acc10[v];
      if (col1 < N) C[(size_t)r1 * N + col1] = acc11[v];
    }
  }
}

// ---------------------------------------------------------------------------
// f32 -> bf16 conversion kernels (weight prep, once per launch)
// ---------------------------------------------------------------------------
__global__ void cvt_bf16_kernel(const float* __restrict__ src, bf16_t* __restrict__ dst, int n) {
  int i = blockIdx.x * blockDim.x + threadIdx.x;
  if (i < n) dst[i] = f2bf(src[i]);
}

// Convert per-layer (R x C) matrices into zero-padded (R2 x C2) bf16.
__global__ void cvt_pad_bf16_kernel(const float* __restrict__ src, bf16_t* __restrict__ dst,
                                    int layers, int R, int C, int R2, int C2) {
  int i = blockIdx.x * blockDim.x + threadIdx.x;
  int total = layers * R2 * C2;
  if (i >= total) return;
  int c = i % C2;
  int r = (i / C2) % R2;
  int l = i / (C2 * R2);
  float v = (r < R && c < C) ? src[((size_t)l * R + r) * C + c] : 0.f;
  dst[i] = f2bf(v);
}

// ---------------------------------------------------------------------------
// Embedding assembly: h0[b,t,:] for t<392 = ctxt@Wpe^T + b + pos[masks_ctxt];
// for t>=392 = mask_token[mask_index] + pos[masks_tgt].
// ---------------------------------------------------------------------------
__global__ void embed_kernel(const float* __restrict__ x_tmp, const float* __restrict__ peb,
                             const float* __restrict__ pos, const float* __restrict__ mtok,
                             const int* __restrict__ mctxt, const int* __restrict__ mtgt,
                             const int* __restrict__ midx, float* __restrict__ res) {
  int i = blockIdx.x * blockDim.x + threadIdx.x;
  if (i >= BATCH_ * L_ * DM_) return;
  int d = i % DM_;
  int t = (i / DM_) % L_;
  int b = i / (DM_ * L_);
  float v;
  if (t < NC_) {
    int p = mctxt[b * NC_ + t];
    v = x_tmp[((size_t)(b * NC_ + t)) * DM_ + d] + peb[d] + pos[(size_t)p * DM_ + d];
  } else {
    int p = mtgt[b * NC_ + (t - NC_)];
    v = mtok[(size_t)midx[0] * DM_ + d] + pos[(size_t)p * DM_ + d];
  }
  res[i] = v;
}

__global__ void add_kernel(float* __restrict__ out, const float* __restrict__ a,
                           const float* b, int n) {
  int i = blockIdx.x * blockDim.x + threadIdx.x;
  if (i < n) out[i] = a[i] + b[i];
}

// ---------------------------------------------------------------------------
// RMSNorm: one block per row; outputs bf16 (GEMM A operand).
// ---------------------------------------------------------------------------
__global__ void rmsnorm_kernel(const float* __restrict__ x, const float* __restrict__ w,
                               bf16_t* __restrict__ out, int cols) {
  int r = blockIdx.x;
  const float* xr = x + (size_t)r * cols;
  __shared__ float red[256];
  int tid = threadIdx.x;
  float s = 0.f;
  for (int c = tid; c < cols; c += 256) { float v = xr[c]; s += v * v; }
  red[tid] = s; __syncthreads();
  for (int o = 128; o > 0; o >>= 1) { if (tid < o) red[tid] += red[tid + o]; __syncthreads(); }
  float sc = rsqrtf(red[0] / (float)cols + 1e-5f);
  for (int c = tid; c < cols; c += 256) out[(size_t)r * cols + c] = f2bf(xr[c] * sc * w[c]);
}

// Final norm over the target half only: row r (0..783) -> (b=r/392, t=392+r%392)
__global__ void final_norm_kernel(const float* __restrict__ hres, const float* __restrict__ w,
                                  bf16_t* __restrict__ out) {
  int r = blockIdx.x;
  int b = r / NC_, t = NC_ + (r % NC_);
  const float* xr = hres + ((size_t)(b * L_ + t)) * DM_;
  __shared__ float red[256];
  int tid = threadIdx.x;
  float s = 0.f;
  for (int c = tid; c < DM_; c += 256) { float v = xr[c]; s += v * v; }
  red[tid] = s; __syncthreads();
  for (int o = 128; o > 0; o >>= 1) { if (tid < o) red[tid] += red[tid + o]; __syncthreads(); }
  float sc = rsqrtf(red[0] / (float)DM_ + 1e-5f);
  for (int c = tid; c < DM_; c += 256) out[(size_t)r * DM_ + c] = f2bf(xr[c] * sc * w[c]);
}

// ---------------------------------------------------------------------------
// Depthwise causal conv (k=4) + SiLU. reverse=1 works in reversed-time domain.
// xp = first DI_ columns of each xz row.
// ---------------------------------------------------------------------------
__global__ void conv_silu_kernel(const float* __restrict__ xz, const float* __restrict__ w,
                                 const float* __restrict__ bias,
                                 float* __restrict__ xc_f32, bf16_t* __restrict__ xc_bf,
                                 int reverse) {
  int i = blockIdx.x * blockDim.x + threadIdx.x;
  if (i >= T_ * DI_) return;
  int c = i % DI_;
  int t = (i / DI_) % L_;
  int b = i / (DI_ * L_);
  float acc = bias[c];
#pragma unroll
  for (int j = 0; j < 4; ++j) {
    int tt = t - 3 + j;
    if (tt < 0) continue;
    int ts = reverse ? (L_ - 1 - tt) : tt;
    acc += w[c * 4 + j] * xz[((size_t)(b * L_ + ts)) * (2 * DI_) + c];
  }
  float s = acc / (1.f + __expf(-acc));   // silu
  xc_f32[i] = s;
  xc_bf[i]  = f2bf(s);
}

// Slice dt (cols 0..47 of the 96-wide dbl buffer) into zero-padded bf16 (rows x 64)
__global__ void dtprep_kernel(const float* __restrict__ dbl, bf16_t* __restrict__ dtpad) {
  int i = blockIdx.x * blockDim.x + threadIdx.x;
  if (i >= T_ * RKP_) return;
  int c = i % RKP_;
  int r = i / RKP_;
  float v = (c < RK_) ? dbl[(size_t)r * XPP_ + c] : 0.f;
  dtpad[i] = f2bf(v);
}

// ---------------------------------------------------------------------------
// Selective scan: one thread per (batch, channel). 16-state recurrence in
// registers, sequential over 784 time steps. y is written in-place over
// deltaRaw (read-before-write per element, same thread -> safe alias).
// ---------------------------------------------------------------------------
__global__ void scan_kernel(const float* deltaRaw,            // (T_, DI_) (aliased with y)
                            const float* __restrict__ dtb,    // (DI_)
                            const float* __restrict__ A_log,  // (DI_, DS_)
                            const float* __restrict__ Dp,     // (DI_)
                            const float* __restrict__ xc,     // (T_, DI_) u (f32)
                            const float* __restrict__ xz,     // (T_, 2*DI_) z at col DI_+c
                            const float* __restrict__ dbl,    // (T_, XPP_) B@48, C@64
                            float* y, int reverse) {
  int gid = blockIdx.x * blockDim.x + threadIdx.x;
  if (gid >= BATCH_ * DI_) return;
  int c = gid % DI_;
  int b = gid / DI_;
  float a[DS_], h[DS_];
#pragma unroll
  for (int n = 0; n < DS_; ++n) { a[n] = -__expf(A_log[c * DS_ + n]); h[n] = 0.f; }
  const float dbias = dtb[c];
  const float Dc = Dp[c];
  for (int t = 0; t < L_; ++t) {
    size_t row = (size_t)(b * L_ + t);
    float x = deltaRaw[row * DI_ + c] + dbias;
    float delta = (x > 20.f) ? x : log1pf(__expf(x));   // softplus
    float u = xc[row * DI_ + c];
    float du = delta * u;
    const float* bc = dbl + row * XPP_;
    float acc = 0.f;
#pragma unroll
    for (int n = 0; n < DS_; ++n) {
      h[n] = __expf(delta * a[n]) * h[n] + du * bc[RK_ + n];
      acc += h[n] * bc[RK_ + DS_ + n];
    }
    float yv = acc + u * Dc;
    int zt = reverse ? (L_ - 1 - t) : t;
    float z = xz[((size_t)(b * L_ + zt)) * (2 * DI_) + DI_ + c];
    yv *= z / (1.f + __expf(-z));       // * silu(z)
    y[row * DI_ + c] = yv;
  }
}

// ysum = y_f + flip_time(y_b), emitted as bf16 for the out_proj GEMM
__global__ void combine_kernel(const float* __restrict__ yf, const float* __restrict__ yb,
                               bf16_t* __restrict__ ysum) {
  int i = blockIdx.x * blockDim.x + threadIdx.x;
  if (i >= T_ * DI_) return;
  int c = i % DI_;
  int t = (i / DI_) % L_;
  int b = i / (DI_ * L_);
  float v = yf[i] + yb[((size_t)(b * L_ + (L_ - 1 - t))) * DI_ + c];
  ysum[i] = f2bf(v);
}

__global__ void bias_out_kernel(const float* __restrict__ x, const float* __restrict__ bias,
                                float* __restrict__ out, int n) {
  int i = blockIdx.x * blockDim.x + threadIdx.x;
  if (i < n) out[i] = x[i] + bias[i % DM_];
}

// ---------------------------------------------------------------------------
// Host orchestration
// ---------------------------------------------------------------------------
extern "C" void kernel_launch(void* const* d_in, const int* in_sizes, int n_in,
                              void* d_out, int out_size, void* d_ws, size_t ws_size,
                              hipStream_t stream) {
  (void)in_sizes; (void)n_in; (void)out_size; (void)ws_size;

  const float* ctxt         = (const float*)d_in[0];
  /* d_in[1] = tgt: unused by reference forward */
  const int*   masks_ctxt   = (const int*)d_in[2];
  const int*   masks_tgt    = (const int*)d_in[3];
  const int*   mask_index   = (const int*)d_in[4];
  const float* pred_embed_w = (const float*)d_in[5];
  const float* pred_embed_b = (const float*)d_in[6];
  const float* pos_embed    = (const float*)d_in[7];
  const float* mask_tokens  = (const float*)d_in[8];
  const float* norm_w       = (const float*)d_in[9];
  const float* in_proj      = (const float*)d_in[10];
  const float* conv_w       = (const float*)d_in[11];
  const float* conv_bias    = (const float*)d_in[12];
  const float* xproj_w      = (const float*)d_in[13];
  const float* dt_w         = (const float*)d_in[14];
  const float* dt_b         = (const float*)d_in[15];
  const float* A_log        = (const float*)d_in[16];
  const float* Dparam       = (const float*)d_in[17];
  const float* conv_w_b     = (const float*)d_in[18];
  const float* conv_bias_b  = (const float*)d_in[19];
  const float* xproj_w_b    = (const float*)d_in[20];
  const float* dt_w_b       = (const float*)d_in[21];
  const float* dt_b_b       = (const float*)d_in[22];
  const float* A_log_b      = (const float*)d_in[23];
  const float* Dparam_b     = (const float*)d_in[24];
  const float* out_proj     = (const float*)d_in[25];
  const float* final_norm_w = (const float*)d_in[26];
  const float* proj_w       = (const float*)d_in[27];
  const float* proj_b       = (const float*)d_in[28];
  float* out = (float*)d_out;

  // ---- workspace layout (bump allocator, 256B aligned) ----
  char* base = (char*)d_ws;
  size_t off = 0;
  auto alloc = [&](size_t bytes) -> void* {
    void* p = base + off;
    off = (off + bytes + 255) & ~(size_t)255;
    return p;
  };
  bf16_t* wpe_bf     = (bf16_t*)alloc((size_t)DM_ * DM_ * 2);
  bf16_t* wproj_bf   = (bf16_t*)alloc((size_t)DM_ * DM_ * 2);
  bf16_t* inproj_bf  = (bf16_t*)alloc((size_t)DEPTH_ * 2 * DI_ * DM_ * 2);
  bf16_t* outproj_bf = (bf16_t*)alloc((size_t)DEPTH_ * DM_ * DI_ * 2);
  bf16_t* xprojf_bf  = (bf16_t*)alloc((size_t)DEPTH_ * XPP_ * DI_ * 2);
  bf16_t* xprojb_bf  = (bf16_t*)alloc((size_t)DEPTH_ * XPP_ * DI_ * 2);
  bf16_t* dtwf_bf    = (bf16_t*)alloc((size_t)DEPTH_ * DI_ * RKP_ * 2);
  bf16_t* dtwb_bf    = (bf16_t*)alloc((size_t)DEPTH_ * DI_ * RKP_ * 2);
  bf16_t* ctxt_bf    = (bf16_t*)alloc((size_t)BATCH_ * NC_ * DM_ * 2);
  float*  x_tmp      = (float*) alloc((size_t)BATCH_ * NC_ * DM_ * 4);   // also out_tmp
  float*  resbuf     = (float*) alloc((size_t)T_ * DM_ * 4);
  float*  hbuf       = (float*) alloc((size_t)T_ * DM_ * 4);
  bf16_t* normed_bf  = (bf16_t*)alloc((size_t)T_ * DM_ * 2);
  float*  xzbuf      = (float*) alloc((size_t)T_ * 2 * DI_ * 4);         // also hres at end
  float*  xcf        = (float*) alloc((size_t)T_ * DI_ * 4);
  float*  xcb        = (float*) alloc((size_t)T_ * DI_ * 4);
  bf16_t* xcf_bf     = (bf16_t*)alloc((size_t)T_ * DI_ * 2);
  bf16_t* xcb_bf     = (bf16_t*)alloc((size_t)T_ * DI_ * 2);
  float*  dblf       = (float*) alloc((size_t)T_ * XPP_ * 4);
  float*  dblb       = (float*) alloc((size_t)T_ * XPP_ * 4);
  bf16_t* dtpadf_bf  = (bf16_t*)alloc((size_t)T_ * RKP_ * 2);
  bf16_t* dtpadb_bf  = (bf16_t*)alloc((size_t)T_ * RKP_ * 2);
  float*  deltaf     = (float*) alloc((size_t)T_ * DI_ * 4);             // deltaRaw then y_f
  float*  deltab     = (float*) alloc((size_t)T_ * DI_ * 4);             // deltaRaw then y_b
  bf16_t* ysum_bf    = (bf16_t*)alloc((size_t)T_ * DI_ * 2);
  bf16_t* fnorm_bf   = (bf16_t*)alloc((size_t)BATCH_ * NC_ * DM_ * 2);
  float*  out_tmp    = x_tmp;   // reuse (x_tmp only needed during embedding)
  float*  hres       = xzbuf;   // reuse (xz not needed after last layer)

  auto grids = [](int n) { return (n + 255) / 256; };
  auto cvt = [&](const float* s, bf16_t* d, int n) {
    cvt_bf16_kernel<<<grids(n), 256, 0, stream>>>(s, d, n);
  };
  auto gemm = [&](const bf16_t* A, const bf16_t* W, float* C, int M, int N, int K) {
    int waves = ((M + 31) / 32) * ((N + 31) / 32);
    gemm_bf16_kernel<<<(waves * 32 + 255) / 256, 256, 0, stream>>>(A, W, C, M, N, K);
  };

  // ---- one-time (per launch) weight conversions ----
  cvt(ctxt, ctxt_bf, BATCH_ * NC_ * DM_);
  cvt(pred_embed_w, wpe_bf, DM_ * DM_);
  cvt(proj_w, wproj_bf, DM_ * DM_);
  cvt(in_proj, inproj_bf, DEPTH_ * 2 * DI_ * DM_);
  cvt(out_proj, outproj_bf, DEPTH_ * DM_ * DI_);
  cvt_pad_bf16_kernel<<<grids(DEPTH_ * XPP_ * DI_), 256, 0, stream>>>(
      xproj_w, xprojf_bf, DEPTH_, XP_, DI_, XPP_, DI_);
  cvt_pad_bf16_kernel<<<grids(DEPTH_ * XPP_ * DI_), 256, 0, stream>>>(
      xproj_w_b, xprojb_bf, DEPTH_, XP_, DI_, XPP_, DI_);
  cvt_pad_bf16_kernel<<<grids(DEPTH_ * DI_ * RKP_), 256, 0, stream>>>(
      dt_w, dtwf_bf, DEPTH_, DI_, RK_, DI_, RKP_);
  cvt_pad_bf16_kernel<<<grids(DEPTH_ * DI_ * RKP_), 256, 0, stream>>>(
      dt_w_b, dtwb_bf, DEPTH_, DI_, RK_, DI_, RKP_);

  // ---- embedding: x = ctxt @ Wpe^T ; h0 assembled into resbuf ----
  gemm(ctxt_bf, wpe_bf, x_tmp, BATCH_ * NC_, DM_, DM_);
  embed_kernel<<<grids(T_ * DM_), 256, 0, stream>>>(
      x_tmp, pred_embed_b, pos_embed, mask_tokens, masks_ctxt, masks_tgt, mask_index, resbuf);

  // ---- layers ----
  for (int i = 0; i < DEPTH_; ++i) {
    if (i > 0)
      add_kernel<<<grids(T_ * DM_), 256, 0, stream>>>(resbuf, resbuf, hbuf, T_ * DM_);
    rmsnorm_kernel<<<T_, 256, 0, stream>>>(resbuf, norm_w + (size_t)i * DM_, normed_bf, DM_);
    gemm(normed_bf, inproj_bf + (size_t)i * 2 * DI_ * DM_, xzbuf, T_, 2 * DI_, DM_);

    conv_silu_kernel<<<grids(T_ * DI_), 256, 0, stream>>>(
        xzbuf, conv_w + (size_t)i * DI_ * 4, conv_bias + (size_t)i * DI_, xcf, xcf_bf, 0);
    conv_silu_kernel<<<grids(T_ * DI_), 256, 0, stream>>>(
        xzbuf, conv_w_b + (size_t)i * DI_ * 4, conv_bias_b + (size_t)i * DI_, xcb, xcb_bf, 1);

    gemm(xcf_bf, xprojf_bf + (size_t)i * XPP_ * DI_, dblf, T_, XPP_, DI_);
    gemm(xcb_bf, xprojb_bf + (size_t)i * XPP_ * DI_, dblb, T_, XPP_, DI_);

    dtprep_kernel<<<grids(T_ * RKP_), 256, 0, stream>>>(dblf, dtpadf_bf);
    dtprep_kernel<<<grids(T_ * RKP_), 256, 0, stream>>>(dblb, dtpadb_bf);

    gemm(dtpadf_bf, dtwf_bf + (size_t)i * DI_ * RKP_, deltaf, T_, DI_, RKP_);
    gemm(dtpadb_bf, dtwb_bf + (size_t)i * DI_ * RKP_, deltab, T_, DI_, RKP_);

    scan_kernel<<<grids(BATCH_ * DI_), 256, 0, stream>>>(
        deltaf, dt_b + (size_t)i * DI_, A_log + (size_t)i * DI_ * DS_,
        Dparam + (size_t)i * DI_, xcf, xzbuf, dblf, deltaf, 0);
    scan_kernel<<<grids(BATCH_ * DI_), 256, 0, stream>>>(
        deltab, dt_b_b + (size_t)i * DI_, A_log_b + (size_t)i * DI_ * DS_,
        Dparam_b + (size_t)i * DI_, xcb, xzbuf, dblb, deltab, 1);

    combine_kernel<<<grids(T_ * DI_), 256, 0, stream>>>(deltaf, deltab, ysum_bf);
    gemm(ysum_bf, outproj_bf + (size_t)i * DM_ * DI_, hbuf, T_, DM_, DI_);
  }

  // ---- final norm (target rows only) + output projection ----
  add_kernel<<<grids(T_ * DM_), 256, 0, stream>>>(hres, resbuf, hbuf, T_ * DM_);
  final_norm_kernel<<<BATCH_ * NC_, 256, 0, stream>>>(hres, final_norm_w, fnorm_bf);
  gemm(fnorm_bf, wproj_bf, out_tmp, BATCH_ * NC_, DM_, DM_);
  bias_out_kernel<<<grids(BATCH_ * NC_ * DM_), 256, 0, stream>>>(
      out_tmp, proj_b, out, BATCH_ * NC_ * DM_);
}
